// FEDformer_69750268887102
// MI455X (gfx1250) — matrix operand order repaired and checked by reading, not compiled, not run on hardware
//
#include <hip/hip_runtime.h>
#include <hip/hip_bf16.h>
#include <math.h>

// ---------------- dims ----------------
constexpr int B_ = 32, L_ = 512, C_ = 7;
constexpr int D_ = 1024, H_ = 8, DFF_ = 4096, NL_ = 4, MM_ = 64, NCLS_ = 2;
constexpr int E_ = D_ / H_;            // 128
constexpr int MTOT = B_ * L_;          // 16384

typedef __bf16 v16bf __attribute__((ext_vector_type(16)));
typedef float  v8f   __attribute__((ext_vector_type(8)));

union Frag { unsigned u[8]; v16bf v; };

__device__ __forceinline__ unsigned short f2bf(float x) {
  unsigned u = __float_as_uint(x);
  u += 0x7fffu + ((u >> 16) & 1u);          // round-to-nearest-even
  return (unsigned short)(u >> 16);
}

__device__ __forceinline__ float gelu_exact(float x) {
  return 0.5f * x * (1.0f + erff(x * 0.7071067811865475f));
}

// async global->LDS copy of 16 bytes (8 bf16), tracked by ASYNCcnt
__device__ __forceinline__ void async_b128(unsigned lds_off, const void* gaddr) {
  asm volatile("global_load_async_to_lds_b128 %0, %1, off"
               :: "v"(lds_off), "v"(gaddr) : "memory");
}

// ---------------- f32 -> bf16 bulk convert (n multiple of 2048) ----------------
__global__ void f32_to_bf16_kernel(const float* __restrict__ in,
                                   unsigned short* __restrict__ out)
{
  size_t i = ((size_t)blockIdx.x * 256 + threadIdx.x) * 8;
  float4 a = *(const float4*)(in + i);
  float4 b = *(const float4*)(in + i + 4);
  uint4 p;
  p.x = (unsigned)f2bf(a.x) | ((unsigned)f2bf(a.y) << 16);
  p.y = (unsigned)f2bf(a.z) | ((unsigned)f2bf(a.w) << 16);
  p.z = (unsigned)f2bf(b.x) | ((unsigned)f2bf(b.y) << 16);
  p.w = (unsigned)f2bf(b.z) | ((unsigned)f2bf(b.w) << 16);
  *(uint4*)(out + i) = p;
}

// ---------------- WMMA GEMM: C[M,N] = A[M,K] @ W[N,K]^T  (+epilogue) ----------------
// A, W already bf16. Triple-buffered async LDS staging, one barrier per K-step.
// EPI: 0=+bias  1=+bias+resid  2=gelu  3=+resid  4=accumulate into C
template<int EPI>
__global__ __launch_bounds__(256)
void gemm_bf16(const unsigned short* __restrict__ A, const unsigned short* __restrict__ W,
               const float* __restrict__ bias, const float* __restrict__ resid,
               float* __restrict__ C, int M, int N, int K, int ldA, int ldW)
{
  // [buf 0..2][A/W][128 rows x 40 (32 k, padded for bank spread)] = 60KB
  __shared__ unsigned short smem[3][2][128 * 40];

  const int tid    = threadIdx.x;
  const int mBlock = blockIdx.y * 128;
  const int nBlock = blockIdx.x * 128;
  const int lane = tid & 31, wid = tid >> 5;
  const int half = lane >> 4, l15 = lane & 15;
  const int mBase = (wid & 3) * 32;        // 4 waves along M (32 rows each)
  const int nBase = (wid >> 2) * 64;       // 2 waves along N (64 cols each)

  v8f acc[2][4];
  #pragma unroll
  for (int a = 0; a < 2; ++a)
    #pragma unroll
    for (int b = 0; b < 4; ++b)
      #pragma unroll
      for (int r = 0; r < 8; ++r) acc[a][b][r] = 0.0f;

  const unsigned short* Ab = A + (size_t)mBlock * ldA;
  const unsigned short* Wb = W + (size_t)nBlock * ldW;

  const unsigned ldsBase = (unsigned)(size_t)(void*)&smem[0][0][0];
  // per-thread source coords: 512 segments of 8 bf16 per matrix, 2 per thread
  const int r0 = tid >> 2, c0 = (tid & 3) * 8;
  const int r1 = (tid + 256) >> 2, c1 = ((tid + 256) & 3) * 8;

  auto issue = [&](int buf, int kb) {
    unsigned ba = ldsBase + (unsigned)buf * (2u * 128u * 40u * 2u);
    unsigned bw = ba + 128u * 40u * 2u;
    async_b128(ba + (unsigned)(r0 * 40 + c0) * 2u, Ab + (size_t)r0 * ldA + kb + c0);
    async_b128(bw + (unsigned)(r0 * 40 + c0) * 2u, Wb + (size_t)r0 * ldW + kb + c0);
    async_b128(ba + (unsigned)(r1 * 40 + c1) * 2u, Ab + (size_t)r1 * ldA + kb + c1);
    async_b128(bw + (unsigned)(r1 * 40 + c1) * 2u, Wb + (size_t)r1 * ldW + kb + c1);
  };

  issue(0, 0);
  const int steps = K / 32;
  int bufC = 0, bufI = 1;
  for (int s = 0; s < steps; ++s) {
    if (s + 1 < steps) {
      issue(bufI, (s + 1) * 32);                    // prefetch next stage
      bufI = (bufI == 2) ? 0 : bufI + 1;
      asm volatile("s_wait_asynccnt 4" ::: "memory"); // current stage retired
    } else {
      asm volatile("s_wait_asynccnt 0" ::: "memory");
    }
    __syncthreads();                                 // single barrier per step

    const unsigned short (*sA)[40] = (const unsigned short (*)[40])&smem[bufC][0][0];
    const unsigned short (*sW)[40] = (const unsigned short (*)[40])&smem[bufC][1][0];

    // preload ALL fragments, then back-to-back WMMAs
    Frag fa[2], fb[4];
    #pragma unroll
    for (int tm = 0; tm < 2; ++tm) {
      int row = mBase + tm * 16 + l15;
      #pragma unroll
      for (int v = 0; v < 8; ++v) {
        int k = ((v & 4) << 2) + ((v & 3) << 1) + half * 8;  // ISA 16-bit A 16x32
        fa[tm].u[v] = *(const unsigned*)&sA[row][k];
      }
    }
    #pragma unroll
    for (int tn = 0; tn < 4; ++tn) {
      int col = nBase + tn * 16 + l15;
      #pragma unroll
      for (int v = 0; v < 8; ++v)                            // ISA 16-bit B 32x16
        fb[tn].u[v] = *(const unsigned*)&sW[col][((v & 4) << 2) + ((v & 3) << 1) + half * 8];
    }
    #pragma unroll
    for (int tn = 0; tn < 4; ++tn) {
      acc[0][tn] = __builtin_amdgcn_wmma_f32_16x16x32_bf16(
          false, fa[0].v, false, fb[tn].v, (short)0, acc[0][tn], false, false);
      acc[1][tn] = __builtin_amdgcn_wmma_f32_16x16x32_bf16(
          false, fa[1].v, false, fb[tn].v, (short)0, acc[1][tn], false, false);
    }
    bufC = (bufC == 2) ? 0 : bufC + 1;
  }

  // epilogue: C/D 16x16 f32 layout (VGPR r -> row r + half*8, lane&15 -> col)
  #pragma unroll
  for (int tm = 0; tm < 2; ++tm)
    #pragma unroll
    for (int tn = 0; tn < 4; ++tn)
      #pragma unroll
      for (int r = 0; r < 8; ++r) {
        int row = mBlock + mBase + tm * 16 + half * 8 + r;
        int col = nBlock + nBase + tn * 16 + l15;
        size_t idx = (size_t)row * N + col;
        float v = acc[tm][tn][r];
        if constexpr (EPI == 0) v += bias[col];
        else if constexpr (EPI == 1) v += bias[col] + resid[idx];
        else if constexpr (EPI == 2) v = gelu_exact(v);
        else if constexpr (EPI == 3) v += resid[idx];
        else if constexpr (EPI == 4) v += C[idx];
        C[idx] = v;
      }
}

// ---------------- embedding: circular conv1d(C->D,k=3) + sinusoidal PE ----------------
__global__ void embed_kernel(const float* __restrict__ xe, const float* __restrict__ tw,
                             float* __restrict__ X)
{
  __shared__ float sx[3 * C_];
  int d = blockIdx.x * 256 + threadIdx.x;
  int l = blockIdx.y, b = blockIdx.z;
  if (threadIdx.x < 3 * C_) {
    int t = threadIdx.x / C_;              // tap 0..2
    int c = threadIdx.x % C_;
    int ls = (l - 1 + t + L_) % L_;        // circular pad
    sx[c * 3 + t] = xe[((size_t)b * L_ + ls) * C_ + c];
  }
  __syncthreads();
  float s = 0.f;
  const float* w = tw + (size_t)d * (C_ * 3);
  #pragma unroll
  for (int j = 0; j < C_ * 3; ++j) s += w[j] * sx[j];
  int i2 = d >> 1;
  float div = expf(-(float)(2 * i2) * 9.210340371976184f / (float)D_); // ln(1e4)
  float ang = (float)l * div;
  float pe = (d & 1) ? cosf(ang) : sinf(ang);
  X[((size_t)b * L_ + l) * D_ + d] = s + pe;
}

// ---------------- series decomp: x - movavg25 (edge-replicated) ----------------
__global__ void decomp_kernel(const float* __restrict__ in, float* __restrict__ out)
{
  int d = blockIdx.x * 256 + threadIdx.x;
  int l = blockIdx.y, b = blockIdx.z;
  const float* base = in + (size_t)b * L_ * D_ + d;
  float s = 0.f;
  #pragma unroll
  for (int k = -12; k <= 12; ++k) {
    int ll = l + k; ll = ll < 0 ? 0 : (ll > L_ - 1 ? L_ - 1 : ll);
    s += base[(size_t)ll * D_];
  }
  out[((size_t)b * L_ + l) * D_ + d] = base[(size_t)l * D_] - s * (1.0f / 25.0f);
}

// ---------------- DFT tables (forward uses modes_index; inverse uses bins 0..63) ----
__global__ void tables_kernel(const int* __restrict__ modes, float* __restrict__ cosT,
                              float* __restrict__ sinT, float* __restrict__ icosT,
                              float* __restrict__ isinT)
{
  int idx = blockIdx.x * 256 + threadIdx.x; // L_*MM_ = 32768
  int l = idx / MM_, m = idx % MM_;
  int k = modes[m];
  int p = (k * l) % L_;
  float ang = 6.283185307179586f * (float)p / (float)L_;
  cosT[idx] = cosf(ang);
  sinT[idx] = -sinf(ang);
  int p2 = (m * l) % L_;
  float ang2 = 6.283185307179586f * (float)p2 / (float)L_;
  float mult = (m == 0) ? 1.0f : 2.0f;
  icosT[(size_t)m * L_ + l] =  mult * (1.0f / L_) * cosf(ang2);
  isinT[(size_t)m * L_ + l] = -mult * (1.0f / L_) * sinf(ang2);
}

// ---------------- Fourier: selected-mode DFT of q[b,h,e,:] ----------------
__global__ void f1_kernel(const float* __restrict__ Q, const float* __restrict__ cosT,
                          const float* __restrict__ sinT, float* __restrict__ selRe,
                          float* __restrict__ selIm)
{
  __shared__ float qrow[L_];
  int bhe = blockIdx.x;                    // (b*H+h)*E+e
  int tid = threadIdx.x;                   // 64
  int e = bhe & (E_ - 1);
  int h = (bhe >> 7) & (H_ - 1);
  int b = bhe >> 10;
  const float* qp = Q + (size_t)b * L_ * D_ + h * E_ + e;
  for (int l = tid; l < L_; l += 64) qrow[l] = qp[(size_t)l * D_];
  __syncthreads();
  int m = tid;
  float sr = 0.f, si = 0.f;
  for (int l = 0; l < L_; ++l) {
    float q = qrow[l];
    sr += q * cosT[l * MM_ + m];
    si += q * sinT[l * MM_ + m];
  }
  selRe[(size_t)bhe * MM_ + m] = sr;
  selIm[(size_t)bhe * MM_ + m] = si;
}

// ---------------- Fourier: complex mode mixing 'bhim,hiom->bhom' ----------------
__global__ void f2_kernel(const float* __restrict__ selRe, const float* __restrict__ selIm,
                          const float* __restrict__ wr, const float* __restrict__ wi,
                          float* __restrict__ mixRe, float* __restrict__ mixIm)
{
  int bho = blockIdx.x;
  int m = threadIdx.x;                     // 64
  int o = bho & (E_ - 1);
  int h = (bho >> 7) & (H_ - 1);
  int b = bho >> 10;
  const float* srp = selRe + (size_t)(b * H_ + h) * E_ * MM_ + m;
  const float* sip = selIm + (size_t)(b * H_ + h) * E_ * MM_ + m;
  const float* wrp = wr + ((size_t)h * E_ * E_ + o) * MM_ + m;
  const float* wip = wi + ((size_t)h * E_ * E_ + o) * MM_ + m;
  float ar = 0.f, ai = 0.f;
  for (int i = 0; i < E_; ++i) {
    float xr = srp[(size_t)i * MM_];
    float xi = sip[(size_t)i * MM_];
    float wrv = wrp[(size_t)i * E_ * MM_];
    float wiv = wip[(size_t)i * E_ * MM_];
    ar += xr * wrv - xi * wiv;
    ai += xr * wiv + xi * wrv;
  }
  mixRe[(size_t)bho * MM_ + m] = ar;
  mixIm[(size_t)bho * MM_ + m] = ai;
}

// ---------------- Fourier: iDFT + flat reshape view [B,H,E,L]->[B,L,D] ----------------
__global__ void f3_kernel(const float* __restrict__ mixRe, const float* __restrict__ mixIm,
                          const float* __restrict__ icosT, const float* __restrict__ isinT,
                          float* __restrict__ out)
{
  __shared__ float mr[MM_], mi[MM_];
  int bhe = blockIdx.x;
  int tid = threadIdx.x;                   // 128
  if (tid < MM_) {
    mr[tid] = mixRe[(size_t)bhe * MM_ + tid];
    mi[tid] = mixIm[(size_t)bhe * MM_ + tid];
  }
  __syncthreads();
  int b = bhe >> 10;
  int he = bhe & 1023;                     // h*E+e
  float* op = out + (size_t)b * L_ * D_ + (size_t)he * L_;
  for (int l = tid; l < L_; l += 128) {
    float s = 0.f;
    #pragma unroll 8
    for (int m = 0; m < MM_; ++m)
      s += mr[m] * icosT[m * L_ + l] + mi[m] * isinT[m * L_ + l];
    op[l] = s;                             // flat index == reshape(B,L,H*E) view
  }
}

// ---------------- my_Layernorm ----------------
__global__ void ln1_kernel(const float* __restrict__ x, const float* __restrict__ lnw,
                           const float* __restrict__ lnb, float* __restrict__ xh)
{
  __shared__ float red[256];
  int bl = blockIdx.x, tid = threadIdx.x;
  const float* xp = x + (size_t)bl * D_;
  float v[4]; float s = 0.f;
  #pragma unroll
  for (int i = 0; i < 4; ++i) { v[i] = xp[tid + i * 256]; s += v[i]; }
  red[tid] = s; __syncthreads();
  for (int o = 128; o > 0; o >>= 1) { if (tid < o) red[tid] += red[tid + o]; __syncthreads(); }
  float mu = red[0] * (1.0f / D_);
  __syncthreads();
  float s2 = 0.f;
  #pragma unroll
  for (int i = 0; i < 4; ++i) { float d = v[i] - mu; s2 += d * d; }
  red[tid] = s2; __syncthreads();
  for (int o = 128; o > 0; o >>= 1) { if (tid < o) red[tid] += red[tid + o]; __syncthreads(); }
  float inv = rsqrtf(red[0] * (1.0f / D_) + 1e-5f);
  float* op = xh + (size_t)bl * D_;
  #pragma unroll
  for (int i = 0; i < 4; ++i) {
    int d = tid + i * 256;
    op[d] = (v[i] - mu) * inv * lnw[d] + lnb[d];
  }
}

__global__ void ln2_kernel(const float* __restrict__ xh, float* __restrict__ colmean)
{
  int d = blockIdx.x * 256 + threadIdx.x;
  int b = blockIdx.y;
  const float* xp = xh + (size_t)b * L_ * D_ + d;
  float s = 0.f;
  for (int l = 0; l < L_; ++l) s += xp[(size_t)l * D_];
  colmean[b * D_ + d] = s * (1.0f / L_);
}

__global__ void ln3_kernel(const float* __restrict__ xh, const float* __restrict__ colmean,
                           float* __restrict__ g)
{
  int d = blockIdx.x * 256 + threadIdx.x;
  int l = blockIdx.y, b = blockIdx.z;
  size_t idx = ((size_t)b * L_ + l) * D_ + d;
  g[idx] = gelu_exact(xh[idx] - colmean[b * D_ + d]);
}

// ---------------- classification head ----------------
__global__ void head_kernel(const float* __restrict__ g, const float* __restrict__ pw,
                            const float* __restrict__ pb, float* __restrict__ out)
{
  __shared__ float red[256];
  int n = blockIdx.x, b = blockIdx.y, tid = threadIdx.x;
  const float* gp = g + (size_t)b * (L_ * D_);
  const float* wp = pw + (size_t)n * (L_ * D_);
  float s = 0.f;
  for (int i = tid; i < L_ * D_; i += 256) s += gp[i] * wp[i];
  red[tid] = s; __syncthreads();
  for (int o = 128; o > 0; o >>= 1) { if (tid < o) red[tid] += red[tid + o]; __syncthreads(); }
  if (tid == 0) out[b * NCLS_ + n] = red[0] + pb[n];
}

// ---------------- host ----------------
extern "C" void kernel_launch(void* const* d_in, const int* in_sizes, int n_in,
                              void* d_out, int out_size, void* d_ws, size_t ws_size,
                              hipStream_t stream)
{
  (void)in_sizes; (void)n_in; (void)out_size; (void)ws_size;
  const float* x_enc = (const float*)d_in[0];
  const float* tw    = (const float*)d_in[1];
  const float* qw    = (const float*)d_in[2];
  const float* qb    = (const float*)d_in[3];
  const float* ow    = (const float*)d_in[4];
  const float* ob    = (const float*)d_in[5];
  const float* wfr   = (const float*)d_in[6];
  const float* wfi   = (const float*)d_in[7];
  const float* c1w   = (const float*)d_in[8];
  const float* c2w   = (const float*)d_in[9];
  const float* lnw   = (const float*)d_in[10];
  const float* lnb   = (const float*)d_in[11];
  const float* pw    = (const float*)d_in[12];
  const float* pb    = (const float*)d_in[13];
  const int*   modes = (const int*)d_in[14];
  float* out = (float*)d_out;

  float* ws = (float*)d_ws;
  size_t off = 0;
  auto alloc = [&](size_t n) { float* p = ws + off; off += n; return p; };
  const size_t BLD = (size_t)B_ * L_ * D_;           // 16777216
  const size_t BHEM = (size_t)B_ * H_ * E_ * MM_;
  float* X    = alloc(BLD);
  float* Qb   = alloc(BLD);
  float* A1   = alloc(BLD);
  float* Y1   = alloc(BLD);
  float* selRe = alloc(BHEM);
  float* selIm = alloc(BHEM);
  float* mixRe = alloc(BHEM);
  float* mixIm = alloc(BHEM);
  float* cosT  = alloc((size_t)L_ * MM_);
  float* sinT  = alloc((size_t)L_ * MM_);
  float* icosT = alloc((size_t)MM_ * L_);
  float* isinT = alloc((size_t)MM_ * L_);
  float* colmean = alloc((size_t)B_ * D_);
  // bf16 buffers (sizes in floats = elems/2)
  unsigned short* qwbf = (unsigned short*)alloc((size_t)NL_ * D_ * D_ / 2);
  unsigned short* owbf = (unsigned short*)alloc((size_t)NL_ * D_ * D_ / 2);
  unsigned short* c1bf = (unsigned short*)alloc((size_t)NL_ * DFF_ * D_ / 2);
  unsigned short* c2bf = (unsigned short*)alloc((size_t)NL_ * D_ * DFF_ / 2);
  unsigned short* ABf  = (unsigned short*)alloc(BLD / 2);
  unsigned short* YBf  = (unsigned short*)alloc(BLD / 2);

  auto conv = [&](const float* src, unsigned short* dst, size_t n) {
    f32_to_bf16_kernel<<<(unsigned)(n / 2048), 256, 0, stream>>>(src, dst);
  };
  auto gemm = [&](int epi, const unsigned short* A, const unsigned short* W,
                  const float* bias, const float* resid, float* Cp,
                  int M, int N, int K, int ldA, int ldW) {
    dim3 g(N / 128, M / 128);
    switch (epi) {
      case 0: gemm_bf16<0><<<g, 256, 0, stream>>>(A, W, bias, resid, Cp, M, N, K, ldA, ldW); break;
      case 1: gemm_bf16<1><<<g, 256, 0, stream>>>(A, W, bias, resid, Cp, M, N, K, ldA, ldW); break;
      case 2: gemm_bf16<2><<<g, 256, 0, stream>>>(A, W, bias, resid, Cp, M, N, K, ldA, ldW); break;
      case 3: gemm_bf16<3><<<g, 256, 0, stream>>>(A, W, bias, resid, Cp, M, N, K, ldA, ldW); break;
      default: gemm_bf16<4><<<g, 256, 0, stream>>>(A, W, bias, resid, Cp, M, N, K, ldA, ldW); break;
    }
  };

  // one-time (per launch) weight conversion to bf16
  conv(qw,  qwbf, (size_t)NL_ * D_ * D_);
  conv(ow,  owbf, (size_t)NL_ * D_ * D_);
  conv(c1w, c1bf, (size_t)NL_ * DFF_ * D_);
  conv(c2w, c2bf, (size_t)NL_ * D_ * DFF_);

  dim3 eltGrid(D_ / 256, L_, B_);
  tables_kernel<<<128, 256, 0, stream>>>(modes, cosT, sinT, icosT, isinT);
  embed_kernel<<<eltGrid, 256, 0, stream>>>(x_enc, tw, X);

  for (int i = 0; i < NL_; ++i) {
    const unsigned short* qwbf_i = qwbf + (size_t)i * D_ * D_;
    const unsigned short* owbf_i = owbf + (size_t)i * D_ * D_;
    const unsigned short* c1bf_i = c1bf + (size_t)i * DFF_ * D_;
    const unsigned short* c2bf_i = c2bf + (size_t)i * D_ * DFF_;
    const float* qb_i = qb + (size_t)i * D_;
    const float* ob_i = ob + (size_t)i * D_;
    const float* wr_i = wfr + (size_t)i * H_ * E_ * E_ * MM_;
    const float* wi_i = wfi + (size_t)i * H_ * E_ * E_ * MM_;

    // q = x @ qw^T + qb
    conv(X, ABf, BLD);
    gemm(0, ABf, qwbf_i, qb_i, nullptr, Qb, MTOT, D_, D_, D_, D_);
    // Fourier block
    f1_kernel<<<B_ * H_ * E_, 64, 0, stream>>>(Qb, cosT, sinT, selRe, selIm);
    f2_kernel<<<B_ * H_ * E_, 64, 0, stream>>>(selRe, selIm, wr_i, wi_i, mixRe, mixIm);
    f3_kernel<<<B_ * H_ * E_, 128, 0, stream>>>(mixRe, mixIm, icosT, isinT, A1);
    // x = x + attn @ ow^T + ob
    conv(A1, ABf, BLD);
    gemm(1, ABf, owbf_i, ob_i, X, Qb, MTOT, D_, D_, D_, D_);
    // series decomp -> x2 (A1)
    decomp_kernel<<<eltGrid, 256, 0, stream>>>(Qb, A1);
    // FFN, chunked over DFF (4 x 1024): Qb = x2 + gelu(x2@c1^T)@c2^T
    conv(A1, ABf, BLD);
    for (int ch = 0; ch < 4; ++ch) {
      gemm(2, ABf, c1bf_i + (size_t)ch * 1024 * D_, nullptr, nullptr,
           Y1, MTOT, 1024, D_, D_, D_);
      conv(Y1, YBf, BLD);
      gemm(ch == 0 ? 3 : 4, YBf, c2bf_i + ch * 1024, nullptr,
           (ch == 0 ? A1 : nullptr), Qb, MTOT, D_, 1024, 1024, DFF_);
    }
    // series decomp -> next x
    decomp_kernel<<<eltGrid, 256, 0, stream>>>(Qb, X);
  }

  // my_Layernorm + time-mean subtraction + gelu + head
  ln1_kernel<<<MTOT, 256, 0, stream>>>(X, lnw, lnb, Qb);
  ln2_kernel<<<dim3(D_ / 256, B_), 256, 0, stream>>>(Qb, colmean);
  ln3_kernel<<<eltGrid, 256, 0, stream>>>(Qb, colmean, A1);
  head_kernel<<<dim3(NCLS_, B_), 256, 0, stream>>>(A1, pw, pb, out);
}
